// ContextRelation_Module_77661598646591
// MI455X (gfx1250) — compile-verified, matrix operand order checked by reading, not run on hardware
//
#include <hip/hip_runtime.h>

typedef __attribute__((ext_vector_type(16))) __bf16 v16bf;
typedef __attribute__((ext_vector_type(8)))  float  v8f;

#define TILE_M 128
#define TILE_N 64
#define TILE_K 32
#define LDS_STRIDE 36          // bf16 elements per LDS row (32 data + 4 pad)
#define LDS_STRIDE_DW 18       // dwords per LDS row

// ---------------------------------------------------------------------------
// Fold inference BN (gamma,beta,mean,var stacked [4,N]) into conv weight/bias.
//   scale = g/sqrt(var+eps);  W'[n,c] = W[n,c]*scale;  b' = b*scale + beta - mean*scale
// ---------------------------------------------------------------------------
__global__ void fold_wb_kernel(const float* __restrict__ W, const float* __restrict__ b,
                               const float* __restrict__ bn, int N, int C,
                               __bf16* __restrict__ Wout, float* __restrict__ bout) {
  int n = blockIdx.x;
  float g   = bn[n];
  float be  = bn[N + n];
  float mu  = bn[2 * N + n];
  float var = bn[3 * N + n];
  float scale = g * rsqrtf(var + 1e-5f);
  float shift = be - mu * scale;
  if (threadIdx.x == 0) bout[n] = b[n] * scale + shift;
  for (int c = threadIdx.x; c < C; c += blockDim.x)
    Wout[(size_t)n * C + c] = (__bf16)(W[(size_t)n * C + c] * scale);
}

// ---------------------------------------------------------------------------
// WMMA GEMM with fused bias+ReLU epilogue:  O = relu(A @ W^T + bias)
//   A_F32_CM:  A is f32, channel-major [B, C, S]; row m = b*S + s
//   else:      A is bf16 row-major [M, C]
//   OUT_F32_CM: write f32 channel-major [B, Ntot, S] (final output)
//   else:       write bf16 row-major [M, Ntot]
//   S is a compile-time constant so strided addressing strength-reduces to
//   shifts + immediate IOFFSETs (no v_mul_u64 chains per load).
// Block = 256 threads = 8 waves; tile 128(M) x 64(N); each wave owns 32x32.
// Register-staged pipeline: prefetch next K-chunk during WMMA on current one.
// OOB rows are clamped at load (no divergence) and masked at the epilogue.
// ---------------------------------------------------------------------------
template<bool A_F32_CM, bool OUT_F32_CM, int S>
__global__ __launch_bounds__(256)
void gemm_bnrelu_kernel(const void* __restrict__ Ain,
                        const __bf16* __restrict__ W,
                        const float* __restrict__ bias,
                        void* __restrict__ Oout,
                        int C, int Ntot, int Mtot) {
  __shared__ __bf16 lA[TILE_M * LDS_STRIDE];
  __shared__ __bf16 lB[TILE_N * LDS_STRIDE];
  unsigned int* lA32 = (unsigned int*)lA;
  unsigned int* lB32 = (unsigned int*)lB;

  const int tid  = threadIdx.x;
  const int lane = tid & 31;
  const int lh   = lane & 15;   // row within 16 (M for A-frag, N for B-frag)
  const int hs   = lane >> 4;   // K-half selector
  const int wave = tid >> 5;    // 0..7
  const int wm   = wave & 3;    // 32-row band within tile
  const int wn   = wave >> 2;   // 32-col band within tile

  const int m0 = blockIdx.x * TILE_M;
  const int n0 = blockIdx.y * TILE_N;

  // ---- A-tile loading assignment: 128 rows x 32 k, 256 threads ----
  const int a_m  = tid & 127;          // local row
  const int a_kb = (tid >> 7) * 16;    // local k base (0 or 16)
  const int mg   = m0 + a_m;
  const int mg_l = mg < Mtot ? mg : (Mtot - 1);  // clamp: always-legal loads
  const float*  Af = nullptr;
  const __bf16* Ab = nullptr;
  if constexpr (A_F32_CM) {
    int bidx = mg_l / S;               // S is a compile-time constant
    int s    = mg_l - bidx * S;
    Af = (const float*)Ain + (size_t)bidx * C * S + s + (size_t)a_kb * S;
  } else {
    Ab = (const __bf16*)Ain + (size_t)mg_l * C;
  }

  // ---- B-tile loading assignment: 64 rows(n) x 32 k ----
  const int b_n  = tid & 63;
  const int b_kb = (tid >> 6) * 8;
  const __bf16* Wrow = W + (size_t)(n0 + b_n) * C;

  v8f acc[2][2];
  v8f zero = {0.f, 0.f, 0.f, 0.f, 0.f, 0.f, 0.f, 0.f};
#pragma unroll
  for (int i = 0; i < 2; i++)
#pragma unroll
    for (int j = 0; j < 2; j++) acc[i][j] = zero;

  // staging registers for the pipelined global->LDS path
  float        aF[16];
  unsigned int aU[8];
  unsigned int bU[4];

  auto loadTiles = [&](int k0) {
    if constexpr (A_F32_CM) {
      const float* p = Af + (size_t)k0 * S;  // shift/const-mul, hoisted
#pragma unroll
      for (int j = 0; j < 16; j++)
        aF[j] = p[(size_t)j * S];            // j*S*4 folds into IOFFSET
    } else {
      const unsigned int* src = (const unsigned int*)(Ab + k0 + a_kb);
#pragma unroll
      for (int j = 0; j < 8; j++) aU[j] = src[j];
    }
    const unsigned int* wsrc = (const unsigned int*)(Wrow + k0 + b_kb);
#pragma unroll
    for (int j = 0; j < 4; j++) bU[j] = wsrc[j];
  };

  auto storeTiles = [&]() {
    if constexpr (A_F32_CM) {
#pragma unroll
      for (int j = 0; j < 16; j += 2) {
        union { __bf16 h[2]; unsigned int u; } p;
        p.h[0] = (__bf16)aF[j];
        p.h[1] = (__bf16)aF[j + 1];
        lA32[a_m * LDS_STRIDE_DW + ((a_kb + j) >> 1)] = p.u;
      }
    } else {
#pragma unroll
      for (int j = 0; j < 8; j++)
        lA32[a_m * LDS_STRIDE_DW + (a_kb >> 1) + j] = aU[j];
    }
#pragma unroll
    for (int j = 0; j < 4; j++)
      lB32[b_n * LDS_STRIDE_DW + (b_kb >> 1) + j] = bU[j];
  };

  loadTiles(0);  // prologue prefetch

  for (int k0 = 0; k0 < C; k0 += TILE_K) {
    storeTiles();
    __syncthreads();

    // prefetch next K-chunk; its latency overlaps the WMMA block below
    if (k0 + TILE_K < C) loadTiles(k0 + TILE_K);

    // gather fragments per ISA 16-bit 16x32 layout:
    //   dword p of lane -> K = (p>>2)*16 + hs*8 + 2*(p&3)
    v16bf afrag[2], bfrag[2];
#pragma unroll
    for (int mi = 0; mi < 2; mi++) {
      int row = wm * 32 + mi * 16 + lh;
      unsigned int* dst = (unsigned int*)&afrag[mi];
#pragma unroll
      for (int p = 0; p < 8; p++)
        dst[p] = lA32[row * LDS_STRIDE_DW + (p >> 2) * 8 + hs * 4 + (p & 3)];
    }
#pragma unroll
    for (int ni = 0; ni < 2; ni++) {
      int row = wn * 32 + ni * 16 + lh;
      unsigned int* dst = (unsigned int*)&bfrag[ni];
#pragma unroll
      for (int p = 0; p < 8; p++)
        dst[p] = lB32[row * LDS_STRIDE_DW + (p >> 2) * 8 + hs * 4 + (p & 3)];
    }

#pragma unroll
    for (int mi = 0; mi < 2; mi++)
#pragma unroll
      for (int ni = 0; ni < 2; ni++)
        acc[mi][ni] = __builtin_amdgcn_wmma_f32_16x16x32_bf16(
            false, afrag[mi], false, bfrag[ni], (short)0, acc[mi][ni],
            false, false);
    __syncthreads();
  }

  // ---- epilogue: bias + ReLU, write out ----
  // D layout: element i of v8f -> M = i + 8*hs, N = lh
#pragma unroll
  for (int mi = 0; mi < 2; mi++) {
#pragma unroll
    for (int ni = 0; ni < 2; ni++) {
      int n = n0 + wn * 32 + ni * 16 + lh;
      float bv = bias[n];
#pragma unroll
      for (int i = 0; i < 8; i++) {
        int m = m0 + wm * 32 + mi * 16 + i + 8 * hs;
        if (m >= Mtot) continue;
        float v = acc[mi][ni][i] + bv;
        v = v > 0.f ? v : 0.f;
        if constexpr (OUT_F32_CM) {
          int bidx = m / S;            // compile-time S -> shift
          int s    = m - bidx * S;
          ((float*)Oout)[((size_t)bidx * Ntot + n) * S + s] = v;
        } else {
          ((__bf16*)Oout)[(size_t)m * Ntot + n] = (__bf16)v;
        }
      }
    }
  }
}

// ---------------------------------------------------------------------------
// Fused attention: sim = (q . k) / 16, softmax over Nc=19, ctx = att @ v.
// q:   [B*HW, 256] bf16 row-major
// k,v: [B*19, 256] bf16 row-major (row = b*19 + j, channels contiguous)
// ctx: [B*HW, 256] bf16 row-major
// One thread per spatial row; k/v cached in LDS as packed dwords (broadcast
// reads). bf16->f32 unpack via shifts (exact), f32->bf16 pack via cvt_pk.
// ---------------------------------------------------------------------------
__global__ __launch_bounds__(256)
void attention_kernel(const __bf16* __restrict__ q,
                      const __bf16* __restrict__ kmat,
                      const __bf16* __restrict__ vmat,
                      __bf16* __restrict__ ctx,
                      int HW) {
  const int KC = 256, NC = 19, KC2 = KC / 2;
  __shared__ unsigned int sk[NC * KC2];
  __shared__ unsigned int sv[NC * KC2];

  const int m = blockIdx.x * blockDim.x + threadIdx.x;
  const int b = (blockIdx.x * (int)blockDim.x) / HW;  // 256 | HW -> uniform b

  const unsigned int* ksrc = (const unsigned int*)(kmat + (size_t)b * NC * KC);
  const unsigned int* vsrc = (const unsigned int*)(vmat + (size_t)b * NC * KC);
  for (int i = threadIdx.x; i < NC * KC2; i += blockDim.x) {
    sk[i] = ksrc[i];
    sv[i] = vsrc[i];
  }
  __syncthreads();

  float dot[NC];
#pragma unroll
  for (int j = 0; j < NC; j++) dot[j] = 0.f;

  const unsigned int* q32 = (const unsigned int*)(q + (size_t)m * KC);
  for (int c2 = 0; c2 < KC2; c2++) {
    unsigned int u = q32[c2];
    float qlo = __uint_as_float(u << 16);
    float qhi = __uint_as_float(u & 0xffff0000u);
#pragma unroll
    for (int j = 0; j < NC; j++) {
      unsigned int kv = sk[j * KC2 + c2];
      dot[j] += qlo * __uint_as_float(kv << 16);
      dot[j] += qhi * __uint_as_float(kv & 0xffff0000u);
    }
  }

  // softmax over 19 with scale Kc^-0.5 = 1/16
  float mx = -1e30f;
#pragma unroll
  for (int j = 0; j < NC; j++) {
    dot[j] *= 0.0625f;
    mx = dot[j] > mx ? dot[j] : mx;
  }
  float sum = 0.f;
#pragma unroll
  for (int j = 0; j < NC; j++) {
    dot[j] = __expf(dot[j] - mx);
    sum += dot[j];
  }
  float inv = 1.f / sum;

  unsigned int* c32 = (unsigned int*)(ctx + (size_t)m * KC);
  for (int c2 = 0; c2 < KC2; c2++) {
    float alo = 0.f, ahi = 0.f;
#pragma unroll
    for (int j = 0; j < NC; j++) {
      unsigned int vv = sv[j * KC2 + c2];
      alo += dot[j] * __uint_as_float(vv << 16);
      ahi += dot[j] * __uint_as_float(vv & 0xffff0000u);
    }
    union { __bf16 h[2]; unsigned int u; } p;
    p.h[0] = (__bf16)(alo * inv);
    p.h[1] = (__bf16)(ahi * inv);
    c32[c2] = p.u;
  }
}

// ---------------------------------------------------------------------------
extern "C" void kernel_launch(void* const* d_in, const int* in_sizes, int n_in,
                              void* d_out, int out_size, void* d_ws, size_t ws_size,
                              hipStream_t stream) {
  (void)in_sizes; (void)n_in; (void)out_size; (void)ws_size;

  const float* x    = (const float*)d_in[0];
  const float* cctx = (const float*)d_in[1];
  const float* qW1 = (const float*)d_in[2];  const float* qb1 = (const float*)d_in[3];  const float* qbn1 = (const float*)d_in[4];
  const float* qW2 = (const float*)d_in[5];  const float* qb2 = (const float*)d_in[6];  const float* qbn2 = (const float*)d_in[7];
  const float* kW1 = (const float*)d_in[8];  const float* kb1 = (const float*)d_in[9];  const float* kbn1 = (const float*)d_in[10];
  const float* kW2 = (const float*)d_in[11]; const float* kb2 = (const float*)d_in[12]; const float* kbn2 = (const float*)d_in[13];
  const float* vW  = (const float*)d_in[14]; const float* vb  = (const float*)d_in[15]; const float* vbn  = (const float*)d_in[16];
  const float* uW  = (const float*)d_in[17]; const float* ub  = (const float*)d_in[18]; const float* ubn  = (const float*)d_in[19];

  const int B = 8, Cin = 512, K = 256, H = 128, Wd = 128, NC = 19;
  const int HW = H * Wd;       // 16384
  const int M  = B * HW;       // 131072
  const int Mc = B * NC;       // 152

  // ---- carve workspace ----
  char* wp = (char*)d_ws;
  auto alloc = [&](size_t bytes) -> char* {
    char* p = wp;
    wp += (bytes + 255) & ~(size_t)255;
    return p;
  };
  __bf16* wq1 = (__bf16*)alloc((size_t)K * Cin * 2);
  __bf16* wq2 = (__bf16*)alloc((size_t)K * K * 2);
  __bf16* wk1 = (__bf16*)alloc((size_t)K * Cin * 2);
  __bf16* wk2 = (__bf16*)alloc((size_t)K * K * 2);
  __bf16* wv  = (__bf16*)alloc((size_t)K * Cin * 2);
  __bf16* wu  = (__bf16*)alloc((size_t)Cin * K * 2);
  float*  bq1 = (float*)alloc(512 * 4);
  float*  bq2 = (float*)alloc(512 * 4);
  float*  bk1 = (float*)alloc(512 * 4);
  float*  bk2 = (float*)alloc(512 * 4);
  float*  bv  = (float*)alloc(512 * 4);
  float*  bu  = (float*)alloc(512 * 4);
  __bf16* q1   = (__bf16*)alloc((size_t)M * K * 2);
  __bf16* q2   = (__bf16*)alloc((size_t)M * K * 2);
  __bf16* ctxb = (__bf16*)alloc((size_t)M * K * 2);
  __bf16* k1   = (__bf16*)alloc((size_t)256 * K * 2);
  __bf16* k2   = (__bf16*)alloc((size_t)256 * K * 2);
  __bf16* vbuf = (__bf16*)alloc((size_t)256 * K * 2);

  // ---- fold BN into weights/biases (bf16 weights) ----
  fold_wb_kernel<<<K,   256, 0, stream>>>(qW1, qb1, qbn1, K,   Cin, wq1, bq1);
  fold_wb_kernel<<<K,   256, 0, stream>>>(qW2, qb2, qbn2, K,   K,   wq2, bq2);
  fold_wb_kernel<<<K,   256, 0, stream>>>(kW1, kb1, kbn1, K,   Cin, wk1, bk1);
  fold_wb_kernel<<<K,   256, 0, stream>>>(kW2, kb2, kbn2, K,   K,   wk2, bk2);
  fold_wb_kernel<<<K,   256, 0, stream>>>(vW,  vb,  vbn,  K,   Cin, wv,  bv);
  fold_wb_kernel<<<Cin, 256, 0, stream>>>(uW,  ub,  ubn,  Cin, K,   wu,  bu);

  dim3 blk(256);

  // q1 = relu(BN(x @ qW1^T))          [M, 256] bf16
  gemm_bnrelu_kernel<true, false, 16384>
      <<<dim3(M / TILE_M, K / TILE_N), blk, 0, stream>>>(x, wq1, bq1, q1, Cin, K, M);
  // k1 = relu(BN(context @ kW1^T))    [152, 256]
  gemm_bnrelu_kernel<true, false, 19>
      <<<dim3((Mc + TILE_M - 1) / TILE_M, K / TILE_N), blk, 0, stream>>>(cctx, wk1, bk1, k1, Cin, K, Mc);
  // v  = relu(BN(context @ vW^T))     [152, 256]
  gemm_bnrelu_kernel<true, false, 19>
      <<<dim3((Mc + TILE_M - 1) / TILE_M, K / TILE_N), blk, 0, stream>>>(cctx, wv, bv, vbuf, Cin, K, Mc);
  // q2 = relu(BN(q1 @ qW2^T))         [M, 256]
  gemm_bnrelu_kernel<false, false, 16384>
      <<<dim3(M / TILE_M, K / TILE_N), blk, 0, stream>>>(q1, wq2, bq2, q2, K, K, M);
  // k2 = relu(BN(k1 @ kW2^T))         [152, 256]
  gemm_bnrelu_kernel<false, false, 16384>
      <<<dim3((Mc + TILE_M - 1) / TILE_M, K / TILE_N), blk, 0, stream>>>(k1, wk2, bk2, k2, K, K, Mc);

  // attention: softmax((q2 . k2)/16) @ v  -> ctx  [M, 256]
  attention_kernel<<<M / 256, 256, 0, stream>>>(q2, k2, vbuf, ctxb, HW);

  // out = relu(BN(ctx @ uW^T))  f32 channel-major [B, 512, H, W]
  gemm_bnrelu_kernel<false, true, 16384>
      <<<dim3(M / TILE_M, Cin / TILE_N), blk, 0, stream>>>(ctxb, wu, bu, d_out, K, Cin, M);
}
